// MultiHeadAttention_71141838291870
// MI455X (gfx1250) — matrix operand order repaired and checked by reading, not compiled, not run on hardware
//
#include <hip/hip_runtime.h>
#include <hip/hip_bf16.h>

// ---------------------------------------------------------------------------
// Time-aware multi-head attention for MI455X (gfx1250, wave32, WMMA).
// bf16 matmul data path (f32 WMMA accumulate); one-shot f32->bf16 conversion
// for inputs/weights. Projection GEMMs stage K-panels through LDS with
// register-prefetch double buffering (global latency hidden behind WMMAs);
// attention GEMMs read L2-resident bf16 directly with clause-pipelined loads.
// ---------------------------------------------------------------------------

typedef __attribute__((ext_vector_type(16))) __bf16 v16bf;
typedef __attribute__((ext_vector_type(8)))  __bf16 v8bf;   // 16 bytes
typedef __attribute__((ext_vector_type(8)))  float  v8f;

#define BB   32
#define SS   256
#define DD   1024
#define HH   8
#define DPH  128
#define TDIM 32
#define BH   (BB * HH)          // 256
#define MROWS (BB * SS)         // 8192
#define LN_EPS 1e-5f

// LDS tile row stride (bf16 elements): 88*2=176 bytes, multiple of 16 for
// aligned ds_load_b128; 176/4 = 44 banks -> r*44 mod 64 distinct for r=0..15
// (conflict-free fragment reads).
#define LSTR 88

__device__ inline v16bf cat8(v8bf lo, v8bf hi) {
  return __builtin_shufflevector(lo, hi, 0, 1, 2, 3, 4, 5, 6, 7,
                                 8, 9, 10, 11, 12, 13, 14, 15);
}

// A-matrix 16x32 bf16 fragment (ISA 7.12.2): lane L<16: M=L, K=k0+{0..7} then
// {16..23}; lanes 16..31: M=L-16, K=k0+{8..15},{24..31}. Source [row][ld] bf16.
__device__ inline v16bf fragA(const __bf16* __restrict__ base,
                              int row0, int ld, int k0) {
  const int lane = threadIdx.x & 31;
  const __bf16* p = base + (long)(row0 + (lane & 15)) * ld
                         + k0 + ((lane & 16) ? 8 : 0);
  v8bf lo = *(const v8bf*)(p);
  v8bf hi = *(const v8bf*)(p + 16);
  return cat8(lo, hi);
}

// B-matrix 32x16 bf16 fragment (dense-B, ISA 7.12.5 pattern): lane L holds
// column N=n0+(L&15); lanes 0..15: K=k0+{0..7},{16..23}, lanes 16..31: K=k0+{8..15},{24..31}.
// Source stored [N][ld] bf16 row-major.
__device__ inline v16bf fragB(const __bf16* __restrict__ base,
                              int col0, int ld, int k0) {
  const int lane = threadIdx.x & 31;
  const __bf16* p = base + (long)(col0 + (lane & 15)) * ld
                         + k0 + ((lane & 16) ? 8 : 0);
  v8bf lo = *(const v8bf*)(p);
  v8bf hi = *(const v8bf*)(p + 16);
  return cat8(lo, hi);
}

__device__ inline v8f wmma_bf16(v16bf a, v16bf b, v8f c) {
  return __builtin_amdgcn_wmma_f32_16x16x32_bf16(false, a, false, b,
                                                 (short)0, c, false, false);
}

// ---------------------------------------------------------------------------
// f32 -> bf16 bulk convert (n multiple of 2048; 8 elems/thread).
// ---------------------------------------------------------------------------
__global__ __launch_bounds__(256)
void cvt_f32_bf16(const float* __restrict__ in, __bf16* __restrict__ out) {
  const long i = ((long)blockIdx.x * 256 + threadIdx.x) * 8;
  float4 a = *(const float4*)(in + i);
  float4 b = *(const float4*)(in + i + 4);
  v8bf o;
  o[0] = (__bf16)a.x; o[1] = (__bf16)a.y; o[2] = (__bf16)a.z; o[3] = (__bf16)a.w;
  o[4] = (__bf16)b.x; o[5] = (__bf16)b.y; o[6] = (__bf16)b.z; o[7] = (__bf16)b.w;
  *(v8bf*)(out + i) = o;
}

// ---------------------------------------------------------------------------
// NT GEMM, bf16 in, f32 accumulate: C[M,N] = A[M,K] @ W[N,K]^T + bias[N].
// 256 threads = 8 waves; block tile 128x128; 64-wide K panels staged in LDS.
// Register-prefetch double buffering: panel p+1's global loads are issued
// before computing panel p, so their latency is covered by 16 WMMAs + 36
// ds_loads; the wait lands at the next iteration's LDS store.
// ---------------------------------------------------------------------------
template <bool OUTBF>
__global__ __launch_bounds__(256)
void gemm_nt_bias(const __bf16* __restrict__ A, const __bf16* __restrict__ W,
                  const float* __restrict__ bias, float* __restrict__ Cf,
                  __bf16* __restrict__ Cb, int Ndim, int Kdim) {
  __shared__ __bf16 lA[128 * LSTR];
  __shared__ __bf16 lB[128 * LSTR];

  const int nb   = blockIdx.x;
  const int mb   = blockIdx.y;
  const int tid  = threadIdx.x;
  const int wave = tid >> 5;
  const int lane = tid & 31;
  const int m0   = wave * 16;          // within the 128-row block tile

  // Per-thread staging coordinates: 1024 16B chunks per tile, 4 per thread.
  int srow[4], soff[4];
#pragma unroll
  for (int i = 0; i < 4; ++i) {
    const int ch = tid + i * 256;
    srow[i] = ch >> 3;
    soff[i] = (ch & 7) * 8;
  }

  v8f c[8];
#pragma unroll
  for (int t = 0; t < 8; ++t)
#pragma unroll
    for (int e = 0; e < 8; ++e) c[t][e] = 0.0f;

  // Prefetch panel 0 into registers.
  v8bf ra[4], rw[4];
#pragma unroll
  for (int i = 0; i < 4; ++i) {
    ra[i] = *(const v8bf*)(A + (long)(mb * 128 + srow[i]) * Kdim + soff[i]);
    rw[i] = *(const v8bf*)(W + (long)(nb * 128 + srow[i]) * Kdim + soff[i]);
  }

  const int nPanels = Kdim >> 6;       // 64-wide K panels
  for (int p = 0; p < nPanels; ++p) {
    // ---- commit prefetched panel to LDS ----------------------------------
#pragma unroll
    for (int i = 0; i < 4; ++i) {
      *(v8bf*)(lA + srow[i] * LSTR + soff[i]) = ra[i];
      *(v8bf*)(lB + srow[i] * LSTR + soff[i]) = rw[i];
    }
    __syncthreads();

    // ---- issue next panel's global loads (latency hidden by compute) -----
    if (p + 1 < nPanels) {
      const int k0 = (p + 1) << 6;
#pragma unroll
      for (int i = 0; i < 4; ++i) {
        ra[i] = *(const v8bf*)(A + (long)(mb * 128 + srow[i]) * Kdim + k0 + soff[i]);
        rw[i] = *(const v8bf*)(W + (long)(nb * 128 + srow[i]) * Kdim + k0 + soff[i]);
      }
    }

    // ---- compute: 2 WMMA k-steps over the 64-wide panel ------------------
#pragma unroll
    for (int kk = 0; kk < 64; kk += 32) {
      v16bf a = fragA(lA, m0, LSTR, kk);
#pragma unroll
      for (int t = 0; t < 8; ++t) {
        v16bf b = fragB(lB, t * 16, LSTR, kk);
        c[t] = wmma_bf16(a, b, c[t]);
      }
    }
    __syncthreads();
  }

  // ---- epilogue: C layout per ISA (row = r + 8*(lane>=16), col = lane&15) --
  const int cl    = lane & 15;
  const int rbase = (lane & 16) ? 8 : 0;
#pragma unroll
  for (int t = 0; t < 8; ++t) {
    const int col = nb * 128 + t * 16 + cl;
    const float bv = bias[col];
#pragma unroll
    for (int r = 0; r < 8; ++r) {
      const long row = (long)mb * 128 + m0 + r + rbase;
      const float vv = c[t][r] + bv;
      if (OUTBF) Cb[row * Ndim + col] = (__bf16)vv;
      else       Cf[row * Ndim + col] = vv;
    }
  }
}

// ---------------------------------------------------------------------------
// t_bias[b,i,j] = sum_td time_weight[td] * cos((t[b,i]-t[b,j])*w[td]+bias[td])
// ---------------------------------------------------------------------------
__global__ __launch_bounds__(256)
void time_bias_kernel(const float* __restrict__ t_seq,
                      const float* __restrict__ tw,
                      const float* __restrict__ tb,
                      const float* __restrict__ twt,
                      float* __restrict__ out) {
  const int idx = blockIdx.x * 256 + threadIdx.x;   // 2^21
  const int j = idx & 255;
  const int i = (idx >> 8) & 255;
  const int b = idx >> 16;
  const float td = t_seq[b * SS + i] - t_seq[b * SS + j];
  float acc = 0.0f;
#pragma unroll
  for (int t = 0; t < TDIM; ++t)
    acc += twt[t] * __cosf(td * tw[t] + tb[t]);
  out[idx] = acc;
}

// ---------------------------------------------------------------------------
// Raw scores: attn_raw[bh,i,j] = 0.25*(Q_bh K_bh^T)[i,j] + tbias[bh%32][i,j]
// Q,K bf16 [bh][256][128]. Block per (qt,bh); wave w covers key cols
// [w*32, w*32+32); K-loop fully unrolled (4 WMMA steps), direct L2 loads.
// ---------------------------------------------------------------------------
__global__ __launch_bounds__(256)
void attn_scores(const __bf16* __restrict__ qb, const __bf16* __restrict__ kb,
                 const float* __restrict__ tbias, float* __restrict__ attn_raw) {
  const int qt   = blockIdx.x;          // 16
  const int bh   = blockIdx.y;          // 256
  const int wave = threadIdx.x >> 5;
  const int lane = threadIdx.x & 31;
  const __bf16* Q  = qb + (long)bh * SS * DPH;
  const __bf16* Km = kb + (long)bh * SS * DPH;
  const int m0 = qt * 16;
  const int n0 = wave * 32;

  v8f c0, c1;
#pragma unroll
  for (int e = 0; e < 8; ++e) { c0[e] = 0.0f; c1[e] = 0.0f; }

#pragma unroll
  for (int k = 0; k < DPH; k += 32) {
    v16bf a  = fragA(Q, m0, DPH, k);
    v16bf b0 = fragB(Km, n0,      DPH, k);
    v16bf b1 = fragB(Km, n0 + 16, DPH, k);
    c0 = wmma_bf16(a, b0, c0);
    c1 = wmma_bf16(a, b1, c1);
  }

  const float* tbb = tbias + (long)(bh % BB) * SS * SS;
  float* out = attn_raw + (long)bh * SS * SS;
  const int cl    = lane & 15;
  const int rbase = (lane & 16) ? 8 : 0;
#pragma unroll
  for (int r = 0; r < 8; ++r) {
    const int row = m0 + r + rbase;
    out[row * SS + n0 + cl]      = c0[r] * 0.25f + tbb[row * SS + n0 + cl];
    out[row * SS + n0 + 16 + cl] = c1[r] * 0.25f + tbb[row * SS + n0 + 16 + cl];
  }
}

// ---------------------------------------------------------------------------
// Row softmax over 256 cols. Writes f32 (final attn output in d_out) and a
// bf16 copy for the context GEMM. One block per row.
// ---------------------------------------------------------------------------
__global__ __launch_bounds__(256)
void softmax_rows(float* __restrict__ attn, __bf16* __restrict__ attn16) {
  const int row = blockIdx.x;           // 65536
  const int tid = threadIdx.x;
  __shared__ float red[256];
  const float v = attn[(long)row * SS + tid];

  red[tid] = v;
  __syncthreads();
#pragma unroll
  for (int s = 128; s > 0; s >>= 1) {
    if (tid < s) red[tid] = fmaxf(red[tid], red[tid + s]);
    __syncthreads();
  }
  const float mx = red[0];
  __syncthreads();

  const float e = __expf(v - mx);
  red[tid] = e;
  __syncthreads();
#pragma unroll
  for (int s = 128; s > 0; s >>= 1) {
    if (tid < s) red[tid] += red[tid + s];
    __syncthreads();
  }
  const float p = e / red[0];
  attn[(long)row * SS + tid]   = p;
  attn16[(long)row * SS + tid] = (__bf16)p;
}

// ---------------------------------------------------------------------------
// Per-head transpose (bf16): vt[bh][d][j] = v[bh][j][d]
// ---------------------------------------------------------------------------
__global__ __launch_bounds__(256)
void transpose_v(const __bf16* __restrict__ vsrc, __bf16* __restrict__ vt) {
  const int idx = blockIdx.x * 256 + threadIdx.x;   // 2^23
  const int d  = idx & 127;
  const int j  = (idx >> 7) & 255;
  const int bh = idx >> 15;
  vt[(long)bh * DPH * SS + d * SS + j] = vsrc[idx];
}

// ---------------------------------------------------------------------------
// context[bh] = attn16[bh] (256x256) @ V[bh] (256x128), V pre-transposed.
// bf16 output (feeds the output-projection GEMM as its A operand).
// ---------------------------------------------------------------------------
__global__ __launch_bounds__(256)
void attn_context(const __bf16* __restrict__ attn, const __bf16* __restrict__ vt,
                  __bf16* __restrict__ ctx) {
  const int mt   = blockIdx.x;          // 16
  const int bh   = blockIdx.y;          // 256
  const int wave = threadIdx.x >> 5;
  const int lane = threadIdx.x & 31;
  const __bf16* Am = attn + (long)bh * SS * SS;     // [256,256]
  const __bf16* Bm = vt   + (long)bh * DPH * SS;    // [128(d),256(j)]
  const int m0 = mt * 16;
  const int n0 = wave * 16;

  v8f c;
#pragma unroll
  for (int e = 0; e < 8; ++e) c[e] = 0.0f;

#pragma unroll
  for (int k = 0; k < SS; k += 32) {
    v16bf a = fragA(Am, m0, SS, k);
    v16bf b = fragB(Bm, n0, SS, k);
    c = wmma_bf16(a, b, c);
  }

  __bf16* out = ctx + (long)bh * SS * DPH;
  const int cl    = lane & 15;
  const int rbase = (lane & 16) ? 8 : 0;
#pragma unroll
  for (int r = 0; r < 8; ++r)
    out[(m0 + r + rbase) * DPH + n0 + cl] = (__bf16)c[r];
}

// ---------------------------------------------------------------------------
// out = LayerNorm(xb + residual) * gamma + beta ; one block per 1024-row.
// ---------------------------------------------------------------------------
__global__ __launch_bounds__(256)
void residual_ln(const float* __restrict__ xb, const float* __restrict__ resid,
                 const float* __restrict__ gamma, const float* __restrict__ beta,
                 float* __restrict__ out) {
  const int row = blockIdx.x;           // 8192
  const int tid = threadIdx.x;
  __shared__ float s1[256];
  __shared__ float s2[256];

  float xv[4];
  float lsum = 0.0f, lsq = 0.0f;
#pragma unroll
  for (int i = 0; i < 4; ++i) {
    const int col = tid + i * 256;
    const float x = xb[(long)row * DD + col] + resid[(long)row * DD + col];
    xv[i] = x;
    lsum += x;
    lsq  += x * x;
  }
  s1[tid] = lsum; s2[tid] = lsq;
  __syncthreads();
#pragma unroll
  for (int s = 128; s > 0; s >>= 1) {
    if (tid < s) { s1[tid] += s1[tid + s]; s2[tid] += s2[tid + s]; }
    __syncthreads();
  }
  const float mean = s1[0] * (1.0f / DD);
  const float var  = s2[0] * (1.0f / DD) - mean * mean;
  const float inv  = rsqrtf(var + LN_EPS);
#pragma unroll
  for (int i = 0; i < 4; ++i) {
    const int col = tid + i * 256;
    out[(long)row * DD + col] = (xv[i] - mean) * inv * gamma[col] + beta[col];
  }
}

// ---------------------------------------------------------------------------
extern "C" void kernel_launch(void* const* d_in, const int* in_sizes, int n_in,
                              void* d_out, int out_size, void* d_ws, size_t ws_size,
                              hipStream_t stream) {
  (void)in_sizes; (void)n_in; (void)out_size; (void)ws_size;

  const float* query = (const float*)d_in[0];
  const float* key   = (const float*)d_in[1];
  const float* value = (const float*)d_in[2];
  const float* t_seq = (const float*)d_in[3];
  const float* Wq = (const float*)d_in[4];
  const float* bq = (const float*)d_in[5];
  const float* Wk = (const float*)d_in[6];
  const float* bk = (const float*)d_in[7];
  const float* Wv = (const float*)d_in[8];
  const float* bv = (const float*)d_in[9];
  const float* Wo = (const float*)d_in[10];
  const float* bo = (const float*)d_in[11];
  const float* gamma = (const float*)d_in[12];
  const float* beta  = (const float*)d_in[13];
  const float* time_w  = (const float*)d_in[14];
  const float* time_b  = (const float*)d_in[15];
  const float* time_wt = (const float*)d_in[16];

  float* outp     = (float*)d_out;                         // [8192,1024]
  float* attn_out = (float*)d_out + (long)MROWS * DD;      // [256,256,256]

  // --- workspace layout ----------------------------------------------------
  const long PROJ   = (long)MROWS * DD;     // 8,388,608 elements
  const long WELEM  = (long)DD * DD;        // 1,048,576 elements
  char* w = (char*)d_ws;
  __bf16* qx16 = (__bf16*)w;                 w += PROJ * 2;   // bf16 query
  __bf16* kx16 = (__bf16*)w;                 w += PROJ * 2;   // bf16 key
  __bf16* vx16 = (__bf16*)w;                 w += PROJ * 2;   // bf16 value
  __bf16* qb16 = (__bf16*)w;                 w += PROJ * 2;   // proj Q (bf16)
  __bf16* kb16 = (__bf16*)w;                 w += PROJ * 2;   // proj K (bf16)
  __bf16* vb16 = (__bf16*)w;                 w += PROJ * 2;   // proj V (bf16)
  __bf16* wq16 = (__bf16*)w;                 w += WELEM * 2;
  __bf16* wk16 = (__bf16*)w;                 w += WELEM * 2;
  __bf16* wv16 = (__bf16*)w;                 w += WELEM * 2;
  __bf16* wo16 = (__bf16*)w;                 w += WELEM * 2;
  float*  tb   = (float*)w;                  w += (long)BB * SS * SS * 4;
  __bf16* at16 = (__bf16*)w;                 w += (long)BH * SS * SS * 2;
  float*  xb   = (float*)w;                  w += PROJ * 4;
  __bf16* vt16 = qx16;   // retired after Q projection
  __bf16* ct16 = kx16;   // retired after K projection

  dim3 blk(256);

  // 0) one-shot f32 -> bf16 conversions (activations + weights)
  cvt_f32_bf16<<<dim3(PROJ / 2048),  blk, 0, stream>>>(query, qx16);
  cvt_f32_bf16<<<dim3(PROJ / 2048),  blk, 0, stream>>>(key,   kx16);
  cvt_f32_bf16<<<dim3(PROJ / 2048),  blk, 0, stream>>>(value, vx16);
  cvt_f32_bf16<<<dim3(WELEM / 2048), blk, 0, stream>>>(Wq, wq16);
  cvt_f32_bf16<<<dim3(WELEM / 2048), blk, 0, stream>>>(Wk, wk16);
  cvt_f32_bf16<<<dim3(WELEM / 2048), blk, 0, stream>>>(Wv, wv16);
  cvt_f32_bf16<<<dim3(WELEM / 2048), blk, 0, stream>>>(Wo, wo16);

  // 1) Q/K/V projections (LDS-staged, register-prefetch WMMA GEMM, bf16 out)
  dim3 ggrid(DD / 128, MROWS / 128);          // (8, 64)
  gemm_nt_bias<true><<<ggrid, blk, 0, stream>>>(qx16, wq16, bq, nullptr, qb16, DD, DD);
  gemm_nt_bias<true><<<ggrid, blk, 0, stream>>>(kx16, wk16, bk, nullptr, kb16, DD, DD);
  gemm_nt_bias<true><<<ggrid, blk, 0, stream>>>(vx16, wv16, bv, nullptr, vb16, DD, DD);

  // 2) time bias
  time_bias_kernel<<<dim3((BB * SS * SS) / 256), blk, 0, stream>>>(
      t_seq, time_w, time_b, time_wt, tb);

  // 3) raw scores + bias -> attn region of d_out
  attn_scores<<<dim3(16, BH), blk, 0, stream>>>(qb16, kb16, tb, attn_out);

  // 4) softmax in place (f32 output) + bf16 copy for context GEMM
  softmax_rows<<<dim3(BH * SS), blk, 0, stream>>>(attn_out, at16);

  // 5) transpose V per head (into retired qx16 buffer)
  transpose_v<<<dim3((BH * SS * DPH) / 256), blk, 0, stream>>>(vb16, vt16);

  // 6) context = attn @ V (bf16 out, into retired kx16 buffer)
  attn_context<<<dim3(16, BH), blk, 0, stream>>>(at16, vt16, ct16);

  // 7) output projection -> xb (f32)
  gemm_nt_bias<false><<<ggrid, blk, 0, stream>>>(ct16, wo16, bo, xb, nullptr, DD, DD);

  // 8) residual + layernorm -> first region of d_out
  residual_ln<<<dim3(MROWS), blk, 0, stream>>>(xb, query, gamma, beta, outp);
}
